// OGN_63402307223700
// MI455X (gfx1250) — matrix operand order repaired and checked by reading, not compiled, not run on hardware
//
#include <hip/hip_runtime.h>
#include <hip/hip_bf16.h>

typedef __attribute__((ext_vector_type(16))) __bf16 v16bf;
typedef __attribute__((ext_vector_type(8)))  __bf16 v8bf;
typedef __attribute__((ext_vector_type(8)))  float  v8f;
typedef __attribute__((ext_vector_type(4)))  unsigned int v4u;
typedef __attribute__((ext_vector_type(8)))  int v8i;
typedef __attribute__((ext_vector_type(4)))  int v4i;

#define DEVINL __device__ __forceinline__

#if defined(__HIP_DEVICE_COMPILE__) && __has_builtin(__builtin_amdgcn_tensor_load_to_lds)
#define HAVE_TDM 1
#else
#define HAVE_TDM 0
#endif

#if defined(__HIP_DEVICE_COMPILE__) && __has_builtin(__builtin_amdgcn_s_wait_tensorcnt)
#define TDM_WAIT() __builtin_amdgcn_s_wait_tensorcnt(0)
#else
#define TDM_WAIT()
#endif

// ---- Tensor Data Mover: 2-D bf16 tile, global -> LDS (D# per ISA 8.3/8.4) ---
// Issue from one wave only; descriptor is uniform (SGPR) data.
DEVINL void tdm_load_2d(void* ldsDst, const void* gsrc, int rows, int rowElems) {
#if HAVE_TDM
  unsigned lds = (unsigned)(size_t)ldsDst;           // aperture low 32 = LDS addr
  unsigned long long ga = (unsigned long long)(size_t)gsrc;
  v4u g0;
  g0[0] = 1u;                                        // count=1, user descriptor
  g0[1] = lds;                                       // lds_addr
  g0[2] = (unsigned)(ga & 0xffffffffu);              // global_addr[31:0]
  g0[3] = (unsigned)((ga >> 32) & 0x01ffffffu)       // global_addr[56:32]
          | (2u << 30);                              // type = 2 ("image")
  v8i g1;
  g1[0] = (1 << 16);                                 // data_size=1 (2 bytes)
  g1[1] = (rowElems & 0xffff) << 16;                 // tensor_dim0[15:0]
  g1[2] = ((rowElems >> 16) & 0xffff)                // tensor_dim0[31:16]
          | ((rows & 0xffff) << 16);                 // tensor_dim1[15:0]
  g1[3] = ((rows >> 16) & 0xffff)                    // tensor_dim1[31:16]
          | ((rowElems & 0xffff) << 16);             // tile_dim0
  g1[4] = (rows & 0xffff);                           // tile_dim1 (tile_dim2=0)
  g1[5] = rowElems;                                  // tensor_dim0_stride[31:0]
  g1[6] = 0;
  g1[7] = 0;
  v4i gz = {0, 0, 0, 0};
#if __clang_major__ >= 23
  v8i gz8 = {0, 0, 0, 0, 0, 0, 0, 0};
  __builtin_amdgcn_tensor_load_to_lds(g0, g1, gz, gz, gz8, 0);
#else
  __builtin_amdgcn_tensor_load_to_lds(g0, g1, gz, gz, 0);
#endif
#else
  (void)ldsDst; (void)gsrc; (void)rows; (void)rowElems;
#endif
}

// swish via hardware reciprocal: v_exp_f32 (neg folded) + v_add + v_rcp + v_mul.
DEVINL float swishf(float x) {
  float e = __expf(-x);
  return x * __builtin_amdgcn_rcpf(1.f + e);
}

DEVINL void zacc(v8f a[8]) {
#pragma unroll
  for (int i = 0; i < 8; ++i)
#pragma unroll
    for (int t = 0; t < 8; ++t) a[i][t] = 0.f;
}

// ---- A-fragment builders (16x32 bf16 A layout) -------------------------
DEVINL v16bf packA_sc(const __bf16* srow, const float* crow, int kc, int lane) {
  const int m  = lane & 15;
  const int kb = kc * 32 + ((lane >> 4) ? 8 : 0);
  const __bf16* p = srow + m * 128 + kb;
  v8bf lo = *(const v8bf*)(p);
  v8bf hi = *(const v8bf*)(p + 16);
  const float* cp = crow + kb;
  v16bf a;
#pragma unroll
  for (int t = 0; t < 8; ++t) {
    float x = (float)lo[t] + cp[t];
    a[t]     = (__bf16)swishf(x);
    float y = (float)hi[t] + cp[t + 16];
    a[t + 8] = (__bf16)swishf(y);
  }
  return a;
}

DEVINL v16bf packA_act(const __bf16* arow, int kc, int lane) {
  const int m  = lane & 15;
  const int kb = kc * 32 + ((lane >> 4) ? 8 : 0);
  const __bf16* p = arow + m * 128 + kb;
  v8bf lo = *(const v8bf*)(p);
  v8bf hi = *(const v8bf*)(p + 16);
  v16bf a;
#pragma unroll
  for (int t = 0; t < 8; ++t) { a[t] = lo[t]; a[t + 8] = hi[t]; }
  return a;
}

// ---- 8 WMMAs (one A chunk against all 8 N-tiles of a 128x128 weight) ---
DEVINL void wmma_row(v8f acc[8], v16bf a, const __bf16* wt, int kc, int lane) {
  const int n0 = lane & 15;
  const int kk = kc * 32 + (lane >> 4) * 8;
#pragma unroll
  for (int nt = 0; nt < 8; ++nt) {
    v8bf blo = *(const v8bf*)(wt + (nt * 16 + n0) * 128 + kk);
    v8bf bhi = *(const v8bf*)(wt + (nt * 16 + n0) * 128 + kk + 16);
    v16bf b;
#pragma unroll
    for (int t = 0; t < 8; ++t) { b[t] = blo[t]; b[t + 8] = bhi[t]; }
    acc[nt] = __builtin_amdgcn_wmma_f32_16x16x32_bf16(
        false, a, false, b, (short)0, acc[nt], false, false);
  }
}

DEVINL void gemm_sc(v8f acc[8], const __bf16* srow, const float* crow,
                    const __bf16* wt, int lane) {
#pragma unroll
  for (int kc = 0; kc < 4; ++kc) {
    v16bf a = packA_sc(srow, crow, kc, lane);
    wmma_row(acc, a, wt, kc, lane);
  }
}

DEVINL void gemm_act(v8f acc[8], const __bf16* arow, const __bf16* wt, int lane) {
#pragma unroll
  for (int kc = 0; kc < 4; ++kc) {
    v16bf a = packA_act(arow, kc, lane);
    wmma_row(acc, a, wt, kc, lane);
  }
}

// ---- epilogues (C layout: VGPR r, lane l -> M = r + 8*(l>>4), N = l&15) -
DEVINL void epi_store(v8f acc[8], const float* bias, __bf16* act, int lane) {
  const int n0 = lane & 15, mh = (lane >> 4) * 8;
#pragma unroll
  for (int nt = 0; nt < 8; ++nt) {
    const float bn = bias[nt * 16 + n0];
#pragma unroll
    for (int r = 0; r < 8; ++r) {
      float x = acc[nt][r] + bn;
      act[(mh + r) * 128 + nt * 16 + n0] = (__bf16)swishf(x);
    }
  }
}

DEVINL void epi_addsc(v8f acc[8], const __bf16* s2row, const float* c2row,
                      __bf16* act, int lane) {
  const int n0 = lane & 15, mh = (lane >> 4) * 8;
#pragma unroll
  for (int nt = 0; nt < 8; ++nt) {
    const int n = nt * 16 + n0;
    const float cv = c2row[n];
#pragma unroll
    for (int r = 0; r < 8; ++r) {
      const int m = mh + r;
      float x = acc[nt][r] + (float)s2row[m * 128 + n] + cv;
      act[m * 128 + n] = (__bf16)swishf(x);
    }
  }
}

DEVINL void epi_agg(v8f acc[8], const float* bias, float* agg, int lane) {
  const int n0 = lane & 15;
#pragma unroll
  for (int nt = 0; nt < 8; ++nt) {
    const float bn = bias[nt * 16 + n0];
    float s = 0.f;
#pragma unroll
    for (int r = 0; r < 8; ++r) s += swishf(acc[nt][r] + bn);
    atomicAdd(&agg[nt * 16 + n0], s);
  }
}

// ======================================================================
// Fused edge kernel: one block = (graph b, 16-dest tile). 8 waves, each
// owning a 16-source strip, run the per-edge WMMA chain and scatter-add
// into an LDS aggregator (never materializing edge features in HBM).
// ======================================================================
template <bool L2>
__global__ __launch_bounds__(256) void edge_kernel(
    const __bf16* __restrict__ s1b, const float* __restrict__ c1,
    const __bf16* __restrict__ s2b, const float* __restrict__ c2,
    const __bf16* __restrict__ w1t, const float* __restrict__ b1,
    const __bf16* __restrict__ wet, const __bf16* __restrict__ w2t,
    const float* __restrict__ b2, float* __restrict__ aggOut) {
  __shared__ __align__(32) __bf16 lds_w1[128 * 128];
  __shared__ __align__(32) __bf16 lds_we[L2 ? 128 * 128 : 16];
  __shared__ __align__(32) __bf16 lds_w2[L2 ? 128 * 128 : 16];
  __shared__ __align__(32) __bf16 lds_s2[L2 ? 128 * 128 : 16];
  __shared__ __align__(32) float  lds_c1[16 * 128];
  __shared__ __align__(32) float  lds_c2[L2 ? 16 * 128 : 16];
  __shared__ __align__(32) float  lds_b1v[128];
  __shared__ __align__(32) float  lds_b2v[L2 ? 128 : 16];
  __shared__ __align__(32) float  lds_agg[16 * 128];
  __shared__ __align__(32) __bf16 lds_act[L2 ? 8 * 16 * 128 : 16];

  const int tid  = threadIdx.x;
  const int wave = tid >> 5, lane = tid & 31;
  const int b = blockIdx.x >> 4, itile = blockIdx.x & 15;
  const int node0 = b * 256 + itile * 16;  // first dest node of this block

  // Stage transposed bf16 weights ([N][K]) via the Tensor Data Mover when
  // available (wave-0 issues; TENSOR ops are wave-level, EXEC-ignored).
#if HAVE_TDM
  if (tid < 32) {
    tdm_load_2d(lds_w1, w1t, 128, 128);
    if constexpr (L2) {
      tdm_load_2d(lds_we, wet, 128, 128);
      tdm_load_2d(lds_w2, w2t, 128, 128);
    }
    TDM_WAIT();
  }
#else
  for (int c = tid; c < 2048; c += 256) {  // 16-byte chunks
    *(v8bf*)(lds_w1 + c * 8) = *(const v8bf*)(w1t + c * 8);
    if constexpr (L2) {
      *(v8bf*)(lds_we + c * 8) = *(const v8bf*)(wet + c * 8);
      *(v8bf*)(lds_w2 + c * 8) = *(const v8bf*)(w2t + c * 8);
    }
  }
#endif
  for (int c = tid; c < 16 * 128; c += 256) {
    lds_c1[c] = c1[node0 * 128 + c];
    if constexpr (L2) lds_c2[c] = c2[node0 * 128 + c];
    lds_agg[c] = 0.f;
  }
  if (tid < 128) {
    lds_b1v[tid] = b1[tid];
    if constexpr (L2) lds_b2v[tid] = b2[tid];
  }
  __syncthreads();

#pragma unroll 1
  for (int jc = 0; jc < 2; ++jc) {   // two 128-source chunks
    const int jbase = b * 256 + jc * 128;
    if (jc == 0)  // CDNA5 global_prefetch_b8 of next chunk's src features
      __builtin_prefetch(s1b + (size_t)(jbase + 128) * 128, 0, 1);
    if constexpr (L2) {
#if HAVE_TDM
      if (tid < 32) {
        tdm_load_2d(lds_s2, s2b + (size_t)jbase * 128, 128, 128);
        TDM_WAIT();
      }
#else
      for (int c = tid; c < 2048; c += 256)
        *(v8bf*)(lds_s2 + c * 8) =
            *(const v8bf*)(s2b + (size_t)jbase * 128 + c * 8);
#endif
      __syncthreads();
    }
    const __bf16* s1row = s1b + (size_t)(jbase + wave * 16) * 128;
    __bf16* myact = lds_act + (L2 ? wave * 2048 : 0);

#pragma unroll 1
    for (int i = 0; i < 16; ++i) {
      v8f acc[8];
      zacc(acc);
      gemm_sc(acc, s1row, lds_c1 + i * 128, lds_w1, lane);  // 32 WMMA
      if constexpr (!L2) {
        epi_agg(acc, lds_b1v, lds_agg + i * 128, lane);     // agg_e1
      } else {
        epi_store(acc, lds_b1v, myact, lane);               // e1 tile
        zacc(acc);
        gemm_act(acc, myact, lds_we, lane);                 // e1 @ W_e
        epi_addsc(acc, lds_s2 + wave * 16 * 128, lds_c2 + i * 128, myact, lane);
        zacc(acc);
        gemm_act(acc, myact, lds_w2, lane);                 // @ B2
        epi_agg(acc, lds_b2v, lds_agg + i * 128, lane);     // agg_e2
      }
    }
    __syncthreads();
  }
  for (int c = tid; c < 2048; c += 256)
    aggOut[(size_t)node0 * 128 + c] = lds_agg[c];
}

// ======================================================================
// Node/global-level helpers (tiny GEMMs, VALU is plenty)
// ======================================================================
__global__ void lin_k(const float* Xa, int ka, int sa, int ra,
                      const float* Xb, int kb, int sb, int rb,
                      const float* Xc, int kc, int sc, int rc,
                      int kones, int rones,
                      const float* __restrict__ W, const float* bias,
                      float* __restrict__ Y, int M, int act) {
  int gid = blockIdx.x * blockDim.x + threadIdx.x;
  if (gid >= M * 128) return;
  int m = gid >> 7, n = gid & 127;
  float s = bias ? bias[n] : 0.f;
  if (Xa) {
    const float* x = Xa + (size_t)(m >> sa) * ka;
    for (int k = 0; k < ka; ++k) s += x[k] * W[(size_t)(ra + k) * 128 + n];
  }
  if (Xb) {
    const float* x = Xb + (size_t)(m >> sb) * kb;
    for (int k = 0; k < kb; ++k) s += x[k] * W[(size_t)(rb + k) * 128 + n];
  }
  if (Xc) {
    const float* x = Xc + (size_t)(m >> sc) * kc;
    for (int k = 0; k < kc; ++k) s += x[k] * W[(size_t)(rc + k) * 128 + n];
  }
  for (int k = 0; k < kones; ++k) s += W[(size_t)(rones + k) * 128 + n];
  if (act) s = swishf(s);
  Y[gid] = s;
}

__global__ void prep_v0(const float* __restrict__ z,
                        const float* __restrict__ sysP, float* __restrict__ v0) {
  int b = blockIdx.x, i = threadIdx.x;  // 8 x 256
  __shared__ float r0[256], r1[256];
  float q0 = z[b * 1024 + i * 2], q1 = z[b * 1024 + i * 2 + 1];
  r0[i] = q0; r1[i] = q1;
  __syncthreads();
  for (int off = 128; off; off >>= 1) {
    if (i < off) { r0[i] += r0[i + off]; r1[i] += r1[i + off]; }
    __syncthreads();
  }
  float m0 = r0[0] * (1.f / 256.f), m1 = r1[0] * (1.f / 256.f);
  float* row = v0 + (size_t)(b * 256 + i) * 6;
  row[0] = q0 - m0;
  row[1] = q1 - m1;
  row[2] = z[b * 1024 + 512 + i * 2];
  row[3] = z[b * 1024 + 512 + i * 2 + 1];
  row[4] = sysP[(b * 256 + i) * 2];
  row[5] = sysP[(b * 256 + i) * 2 + 1];
}

__global__ void reduce_graph(const float* __restrict__ X, float* __restrict__ Y) {
  int b = blockIdx.x, n = threadIdx.x;  // 8 x 128
  float s = 0.f;
  for (int i = 0; i < 256; ++i) s += X[(size_t)(b * 256 + i) * 128 + n];
  Y[b * 128 + n] = s;
}

__global__ void cvt_bf16(const float* __restrict__ X, __bf16* __restrict__ Y, int n) {
  int g = blockIdx.x * blockDim.x + threadIdx.x;
  if (g < n) Y[g] = (__bf16)X[g];
}

// W is (128 x 128) row-major (k,n) -> Wt[n][k] bf16 (B-fragment friendly)
__global__ void transpose_wt(const float* __restrict__ W, __bf16* __restrict__ Wt) {
  int g = blockIdx.x * blockDim.x + threadIdx.x;  // 16384
  int k = g >> 7, n = g & 127;
  Wt[n * 128 + k] = (__bf16)W[g];
}

__global__ void out_kernel(const float* __restrict__ v2,
                           const float* __restrict__ qw, const float* __restrict__ qb,
                           const float* __restrict__ pw, const float* __restrict__ pb,
                           float* __restrict__ out) {
  int g = blockIdx.x * blockDim.x + threadIdx.x;  // 4096
  int node = g >> 1, d = g & 1;
  int b = node >> 8, i = node & 255;
  const float* v = v2 + (size_t)node * 128;
  float sq = qb[d], sp = pb[d];
  for (int n = 0; n < 128; ++n) {
    sq += v[n] * qw[n * 2 + d];
    sp += v[n] * pw[n * 2 + d];
  }
  out[b * 1024 + i * 2 + d] = sq;
  out[b * 1024 + 512 + i * 2 + d] = sp;
}

// ======================================================================
// Host-side orchestration
// ======================================================================
extern "C" void kernel_launch(void* const* d_in, const int* in_sizes, int n_in,
                              void* d_out, int out_size, void* d_ws, size_t ws_size,
                              hipStream_t stream) {
  (void)in_sizes; (void)n_in; (void)out_size; (void)ws_size;
  // Input leaf order: t, z, sysP, then params flattened (sorted dict keys:
  // layers[0..1]{edge,global,node}{l1,l2}{b,w}, plin{b,w}, qlin{b,w}).
  const float* z    = (const float*)d_in[1];
  const float* sysP = (const float*)d_in[2];
  const float* e1b1 = (const float*)d_in[3];
  const float* e1w1 = (const float*)d_in[4];   // 14x128
  const float* e1b2 = (const float*)d_in[5];
  const float* e1w2 = (const float*)d_in[6];   // 128x128
  const float* g1b1 = (const float*)d_in[7];
  const float* g1w1 = (const float*)d_in[8];   // 257x128
  const float* g1b2 = (const float*)d_in[9];
  const float* g1w2 = (const float*)d_in[10];
  const float* n1b1 = (const float*)d_in[11];
  const float* n1w1 = (const float*)d_in[12];  // 135x128
  const float* n1b2 = (const float*)d_in[13];
  const float* n1w2 = (const float*)d_in[14];
  const float* e2b1 = (const float*)d_in[15];
  const float* e2w1 = (const float*)d_in[16];  // 512x128
  const float* e2b2 = (const float*)d_in[17];
  const float* e2w2 = (const float*)d_in[18];
  const float* n2b1 = (const float*)d_in[23];
  const float* n2w1 = (const float*)d_in[24];  // 384x128
  const float* n2b2 = (const float*)d_in[25];
  const float* n2w2 = (const float*)d_in[26];
  const float* plb  = (const float*)d_in[27];
  const float* plw  = (const float*)d_in[28];
  const float* qlb  = (const float*)d_in[29];
  const float* qlw  = (const float*)d_in[30];

  char* wp = (char*)d_ws;
  auto alloc = [&](size_t bytes) {
    void* p = (void*)wp;
    wp += (bytes + 255) & ~(size_t)255;
    return p;
  };
  float*  v0    = (float*)alloc(2048 * 6 * 4);
  float*  s1    = (float*)alloc(2048 * 128 * 4);
  float*  c1    = (float*)alloc(2048 * 128 * 4);
  float*  s2    = (float*)alloc(2048 * 128 * 4);
  float*  c2    = (float*)alloc(2048 * 128 * 4);
  __bf16* s1b   = (__bf16*)alloc(2048 * 128 * 2);
  __bf16* s2b   = (__bf16*)alloc(2048 * 128 * 2);
  __bf16* w1t   = (__bf16*)alloc(16384 * 2);
  __bf16* wet   = (__bf16*)alloc(16384 * 2);
  __bf16* w2t   = (__bf16*)alloc(16384 * 2);
  float*  aggE1 = (float*)alloc(2048 * 128 * 4);
  float*  aggE2 = (float*)alloc(2048 * 128 * 4);
  float*  Hn    = (float*)alloc(2048 * 128 * 4);
  float*  v1    = (float*)alloc(2048 * 128 * 4);
  float*  v2    = (float*)alloc(2048 * 128 * 4);
  float*  ge1   = (float*)alloc(8 * 128 * 4);
  float*  gn1   = (float*)alloc(8 * 128 * 4);
  float*  Hg    = (float*)alloc(8 * 128 * 4);
  float*  u1    = (float*)alloc(8 * 128 * 4);

  const int GN = 1024;
  const float* NUL = nullptr;

  // ---- Phase A: node features + edge-layer-1 split linears ----
  prep_v0<<<8, 256, 0, stream>>>(z, sysP, v0);
  lin_k<<<GN, 256, 0, stream>>>(v0, 6, 0, 0, NUL, 0, 0, 0, NUL, 0, 0, 0,
                                0, 0, e1w1, nullptr, s1, 2048, 0);
  lin_k<<<GN, 256, 0, stream>>>(v0, 6, 0, 6, NUL, 0, 0, 0, NUL, 0, 0, 0,
                                2, 12, e1w1, e1b1, c1, 2048, 0);
  cvt_bf16<<<GN, 256, 0, stream>>>(s1, s1b, 2048 * 128);
  transpose_wt<<<64, 256, 0, stream>>>(e1w2, w1t);

  // ---- Phase B: layer-1 edges (WMMA, fused aggregation) ----
  edge_kernel<false><<<128, 256, 0, stream>>>(s1b, c1, nullptr, nullptr, w1t,
                                              e1b2, nullptr, nullptr, nullptr,
                                              aggE1);

  // ---- Phase C: node/global layer 1 + layer-2 split linears ----
  reduce_graph<<<8, 128, 0, stream>>>(aggE1, ge1);
  lin_k<<<GN, 256, 0, stream>>>(v0, 6, 0, 0, aggE1, 128, 0, 6, NUL, 0, 0, 0,
                                1, 134, n1w1, n1b1, Hn, 2048, 1);
  lin_k<<<GN, 256, 0, stream>>>(Hn, 128, 0, 0, NUL, 0, 0, 0, NUL, 0, 0, 0,
                                0, 0, n1w2, n1b2, v1, 2048, 1);
  reduce_graph<<<8, 128, 0, stream>>>(v1, gn1);
  lin_k<<<4, 256, 0, stream>>>(ge1, 128, 0, 1, gn1, 128, 0, 129, NUL, 0, 0, 0,
                               1, 0, g1w1, g1b1, Hg, 8, 1);
  lin_k<<<4, 256, 0, stream>>>(Hg, 128, 0, 0, NUL, 0, 0, 0, NUL, 0, 0, 0,
                               0, 0, g1w2, g1b2, u1, 8, 1);
  lin_k<<<GN, 256, 0, stream>>>(v1, 128, 0, 0, NUL, 0, 0, 0, NUL, 0, 0, 0,
                                0, 0, e2w1, nullptr, s2, 2048, 0);
  lin_k<<<GN, 256, 0, stream>>>(v1, 128, 0, 128, u1, 128, 8, 384, NUL, 0, 0, 0,
                                0, 0, e2w1, e2b1, c2, 2048, 0);
  cvt_bf16<<<GN, 256, 0, stream>>>(s2, s2b, 2048 * 128);
  transpose_wt<<<64, 256, 0, stream>>>(e2w1 + 256 * 128, wet);  // W_e slice
  transpose_wt<<<64, 256, 0, stream>>>(e2w2, w2t);

  // ---- Phase D: layer-2 edges (WMMA chain: e1 recompute -> e2 -> agg) ----
  edge_kernel<true><<<128, 256, 0, stream>>>(s1b, c1, s2b, c2, w1t, e1b2, wet,
                                             w2t, e2b2, aggE2);

  // ---- Phase E: node layer 2 + output heads (global2 is dead code) ----
  lin_k<<<GN, 256, 0, stream>>>(v1, 128, 0, 0, aggE2, 128, 0, 128, u1, 128, 8, 256,
                                0, 0, n2w1, n2b1, Hn, 2048, 1);
  lin_k<<<GN, 256, 0, stream>>>(Hn, 128, 0, 0, NUL, 0, 0, 0, NUL, 0, 0, 0,
                                0, 0, n2w2, n2b2, v2, 2048, 1);
  out_kernel<<<16, 256, 0, stream>>>(v2, qlw, qlb, plw, plb, (float*)d_out);
}
